// Attention_40407052320989
// MI455X (gfx1250) — matrix operand order repaired and checked by reading, not compiled, not run on hardware
//
#include <hip/hip_runtime.h>
#include <hip/hip_bf16.h>
#include <cstdint>

// ---------------------------------------------------------------------------
// Problem constants (match reference)
// ---------------------------------------------------------------------------
constexpr int SEQ   = 2048;   // SEQLEN
constexpr int MODEL = 4096;   // DIM
constexpr int NH    = 32;     // N_HEADS
constexpr int NKVH  = 8;      // N_KV_HEADS
constexpr int HDIM  = 128;    // HEAD_DIM
constexpr int NB    = 2;      // BSZ
constexpr float NEG_BIG = -1.0e9f;

typedef __attribute__((ext_vector_type(16))) __bf16   v16bf;
typedef __attribute__((ext_vector_type(8)))  float    v8f;
typedef __attribute__((ext_vector_type(8)))  unsigned v8u;

__device__ __forceinline__ unsigned short f2bf(float x) {
  unsigned u = __builtin_bit_cast(unsigned, x);
  u += 0x7FFFu + ((u >> 16) & 1u);          // round-to-nearest-even
  return (unsigned short)(u >> 16);
}
__device__ __forceinline__ unsigned pk2bf(float a, float b) {
  return (unsigned)f2bf(a) | ((unsigned)f2bf(b) << 16);
}
__device__ __forceinline__ v16bf frag_cast(const v8u& u) {
  return __builtin_bit_cast(v16bf, u);
}

// ---- CDNA5 async global->LDS copy (ASYNCcnt path, ISA 15.18.3 op 98) ------
// VDST = LDS byte address (aperture rule: LDS address is the low 32 bits),
// VADDR = 64-bit global address, GV mode (no SADDR).
__device__ __forceinline__ void async_ld_b128(void* lds_dst, const void* gsrc) {
  asm volatile("global_load_async_to_lds_b128 %0, %1, off"
               :: "v"((unsigned)(uintptr_t)lds_dst), "v"(gsrc)
               : "memory");
}
__device__ __forceinline__ void wait_async0() {
  asm volatile("s_wait_asynccnt 0" ::: "memory");
}

// ---------------------------------------------------------------------------
// Tiled bf16 WMMA GEMM:  C[64x64 tile] = A[M x 4096] * W[4096 x N]
//  MODE 0: A=f32, out=bf16 head-major [b][head][s][128] with RoPE (q,k proj)
//  MODE 1: A=f32, out=bf16 head-major, no RoPE                    (v proj)
//  MODE 2: A=bf16 (async-to-LDS), out=f32 row-major [m][n]        (out proj)
// Block: 128 threads = 4 waves; wave w owns a 32x32 sub-tile (2x2 WMMA frags)
// ---------------------------------------------------------------------------
template <int MODE>
__global__ __launch_bounds__(128)
void gemm_wmma(const float* __restrict__ Af, const unsigned short* __restrict__ Ab,
               const float* __restrict__ W, void* __restrict__ Out, int N,
               const float* __restrict__ cosT, const float* __restrict__ sinT,
               int nheads) {
  constexpr int K = MODEL;
  // 48-short row pad => 96B stride (16B multiple: legal b128 LDS stores)
  __shared__ unsigned short Asl[64][48];   // [m][k]
  __shared__ unsigned short Bsl[64][48];   // [n][k] (transposed W tile)

  const int tid  = threadIdx.x;
  const int n0   = blockIdx.x * 64;
  const int m0   = blockIdx.y * 64;
  const int lane = tid & 31, wave = tid >> 5;
  const int lm   = lane & 15, half = lane >> 4;
  const int wm   = (wave >> 1) * 32, wn = (wave & 1) * 32;

  v8f acc[2][2] = {};

  for (int k0 = 0; k0 < K; k0 += 32) {
    // ---- stage A tile (64 rows x 32 k) into LDS as bf16 ----
    if (MODE == 2) {
      // pure bf16 copy: async DMA straight into LDS (no VGPR round-trip)
#pragma unroll
      for (int i = 0; i < 2; ++i) {
        const int c    = tid + i * 128;     // 256 chunks of 8 bf16 (16B)
        const int row  = c >> 2;
        const int coff = (c & 3) * 8;
        async_ld_b128(&Asl[row][coff],
                      Ab + (size_t)(m0 + row) * K + k0 + coff);
      }
    } else {
#pragma unroll
      for (int i = 0; i < 4; ++i) {
        const int c   = tid + i * 128;      // 512 quad-chunks
        const int row = c >> 3;
        const int cq  = (c & 7) * 4;
        const float4 v = *(const float4*)(Af + (size_t)(m0 + row) * K + k0 + cq);
        uint2 p;
        p.x = pk2bf(v.x, v.y);
        p.y = pk2bf(v.z, v.w);
        *(uint2*)&Asl[row][cq] = p;
      }
    }
    // ---- stage W tile transposed: Bsl[n][k] (f32 -> bf16 convert) ----
#pragma unroll
    for (int i = 0; i < 4; ++i) {
      const int c  = tid + i * 128;
      const int kk = c >> 4;               // 0..31
      const int nq = (c & 15) * 4;
      const float4 w4 = *(const float4*)(W + (size_t)(k0 + kk) * N + n0 + nq);
      Bsl[nq + 0][kk] = f2bf(w4.x);
      Bsl[nq + 1][kk] = f2bf(w4.y);
      Bsl[nq + 2][kk] = f2bf(w4.z);
      Bsl[nq + 3][kk] = f2bf(w4.w);
    }
    if (k0 + 32 < K)  // pull next W tile toward L2 (global_prefetch_b8)
      __builtin_prefetch(W + (size_t)(k0 + 32) * N + n0 + ((tid & 7) << 3), 0, 1);
    if (MODE == 2) wait_async0();          // async LDS writes landed
    __syncthreads();

    // ---- load fragments per ISA 7.12.2 layouts, 4 WMMAs ----
    v16bf afr[2], bfr[2];
#pragma unroll
    for (int s = 0; s < 2; ++s) {
      v8u ua, ub;
#pragma unroll
      for (int v = 0; v < 8; ++v) {
        const int ka = (v < 4 ? 0 : 16) + half * 8 + (v & 3) * 2;  // A layout
        ua[v] = *(const unsigned*)&Asl[wm + s * 16 + lm][ka];
        const int kb_ = (v < 4 ? 0 : 16) + half * 8 + (v & 3) * 2;  // B layout
        ub[v] = *(const unsigned*)&Bsl[wn + s * 16 + lm][kb_];
      }
      afr[s] = frag_cast(ua);
      bfr[s] = frag_cast(ub);
    }
#pragma unroll
    for (int i = 0; i < 2; ++i)
#pragma unroll
      for (int j = 0; j < 2; ++j)
        acc[i][j] = __builtin_amdgcn_wmma_f32_16x16x32_bf16(
            false, afr[i], false, bfr[j], (short)0, acc[i][j], false, false);
    __syncthreads();
  }

  // ---- epilogue ----
#pragma unroll
  for (int sm = 0; sm < 2; ++sm)
#pragma unroll
    for (int sn = 0; sn < 2; ++sn)
#pragma unroll
      for (int r = 0; r < 8; ++r) {
        const int m = m0 + wm + sm * 16 + half * 8 + r;   // C frag row map
        const int n = n0 + wn + sn * 16 + lm;             // C frag col map
        const float val = acc[sm][sn][r];
        if (MODE == 2) {
          ((float*)Out)[(size_t)m * N + n] = val;
        } else {
          const int b = m >> 11, sq = m & (SEQ - 1);
          const int head = n >> 7, d = n & 127;
          float o = val;
          if (MODE == 0) {  // RoPE: dim pairs (2i,2i+1) sit on adjacent lanes
            const float p   = __shfl_xor(val, 1, 32);
            const int   fi  = d >> 1;
            const float c   = cosT[sq * (HDIM / 2) + fi];
            const float sn_ = sinT[sq * (HDIM / 2) + fi];
            o = (d & 1) ? (p * sn_ + val * c) : (val * c - p * sn_);
          }
          ((unsigned short*)Out)[((((size_t)b * nheads + head) * SEQ) + sq) * HDIM + d] =
              f2bf(o);
        }
      }
}

// ---------------------------------------------------------------------------
// Flash attention (causal, GQA 4:1), bf16 WMMA with f32 accumulation.
// Grid: (SEQ/64, NH, NB); 128 threads = 4 waves; wave owns 16 q rows.
// K tile staged with async global->LDS DMA; V transposed via VALU path.
// ---------------------------------------------------------------------------
__global__ __launch_bounds__(128)
void flash_attn(const unsigned short* __restrict__ qb,
                const unsigned short* __restrict__ kb,
                const unsigned short* __restrict__ vb,
                unsigned short* __restrict__ attb) {
  __shared__ unsigned short Ks[32][136];     // [pos][d]; 272B stride (16B mult)
  __shared__ unsigned short Vt[HDIM][40];    // V tile transposed [d][pos]
  __shared__ unsigned short Ps[4][16][40];   // per-wave P staging (C->A shape)

  const int tid  = threadIdx.x;
  const int lane = tid & 31, wave = tid >> 5;
  const int lm   = lane & 15, half = lane >> 4;
  const int qblk = blockIdx.x;
  const int h    = blockIdx.y;
  const int b    = blockIdx.z;
  const int kvh  = h >> 2;  // N_REP = 4

  // ---- Q fragments: 16 rows x 128 dims, loaded once ----
  const int qrow = qblk * 64 + wave * 16 + lm;
  const unsigned short* qrp =
      qb + ((((size_t)b * NH + h) * SEQ) + qrow) * HDIM;
  v16bf qf[4];
#pragma unroll
  for (int c = 0; c < 4; ++c) {
    v8u u;
#pragma unroll
    for (int v = 0; v < 8; ++v) {
      const int kk = c * 32 + (v < 4 ? 0 : 16) + half * 8 + (v & 3) * 2;
      u[v] = *(const unsigned*)(qrp + kk);
    }
    qf[c] = frag_cast(u);
  }

  v8f oacc[8] = {};
  float mrow[8], lrow[8];
#pragma unroll
  for (int r = 0; r < 8; ++r) { mrow[r] = -1.0e30f; lrow[r] = 0.f; }

  const int    rowbase = qblk * 64 + wave * 16 + half * 8;
  const float  scale   = 0.088388347648318447f;  // 1/sqrt(128)
  const size_t kvbase  = (((size_t)b * NKVH + kvh) * SEQ) * HDIM;
  const int    jmax    = qblk * 2 + 2;  // uniform across block (causal bound)

  for (int j = 0; j < jmax; ++j) {
    const int p0 = j * 32;
    // ---- stage K (pure bf16 copy, 32 x 128): async DMA into LDS ----
#pragma unroll
    for (int i = 0; i < 4; ++i) {
      const int c    = tid + i * 128;     // 512 chunks of 8 bf16 (16B)
      const int pos  = c >> 4;
      const int doff = (c & 15) * 8;
      async_ld_b128(&Ks[pos][doff],
                    kb + kvbase + (size_t)(p0 + pos) * HDIM + doff);
    }
    // ---- stage V transposed: Vt[d][pos] (overlaps with async K copy) ----
#pragma unroll
    for (int i = 0; i < 16; ++i) {
      const int c = tid + i * 128;
      const int pos = c >> 6;
      const int d2 = (c & 63) * 2;
      const unsigned u =
          *(const unsigned*)(vb + kvbase + (size_t)(p0 + pos) * HDIM + d2);
      Vt[d2][pos]     = (unsigned short)(u & 0xFFFFu);
      Vt[d2 + 1][pos] = (unsigned short)(u >> 16);
    }
    wait_async0();
    __syncthreads();

    // ---- scores = Q * K^T over 2 subtiles of 16 kv cols ----
    v8f st[2] = {};
#pragma unroll
    for (int sub = 0; sub < 2; ++sub) {
#pragma unroll
      for (int c = 0; c < 4; ++c) {
        v8u u;
#pragma unroll
        for (int v = 0; v < 8; ++v) {
          const int kk = c * 32 + (v < 4 ? 0 : 16) + half * 8 + (v & 3) * 2;
          u[v] = *(const unsigned*)&Ks[sub * 16 + lm][kk];
        }
        st[sub] = __builtin_amdgcn_wmma_f32_16x16x32_bf16(
            false, qf[c], false, frag_cast(u), (short)0, st[sub], false, false);
      }
    }

    // ---- causal mask + online softmax (row = r + half*8 per C layout) ----
#pragma unroll
    for (int r = 0; r < 8; ++r) {
      const int qr = rowbase + r;
      float s0 = st[0][r] * scale;
      float s1 = st[1][r] * scale;
      if (p0 + lm > qr)      s0 = NEG_BIG;
      if (p0 + 16 + lm > qr) s1 = NEG_BIG;
      float mx = fmaxf(s0, s1);
#pragma unroll
      for (int off = 1; off < 16; off <<= 1)
        mx = fmaxf(mx, __shfl_xor(mx, off, 32));
      const float mnew  = fmaxf(mrow[r], mx);
      const float alpha = __expf(mrow[r] - mnew);
      const float e0    = __expf(s0 - mnew);
      const float e1    = __expf(s1 - mnew);
      float rs = e0 + e1;
#pragma unroll
      for (int off = 1; off < 16; off <<= 1)
        rs += __shfl_xor(rs, off, 32);
      lrow[r] = lrow[r] * alpha + rs;
      mrow[r] = mnew;
#pragma unroll
      for (int dsub = 0; dsub < 8; ++dsub) oacc[dsub][r] *= alpha;
      Ps[wave][half * 8 + r][lm]      = f2bf(e0);  // reshape C -> A via LDS
      Ps[wave][half * 8 + r][16 + lm] = f2bf(e1);
    }
    __syncthreads();

    // ---- O += P * V ----
    v8u up;
#pragma unroll
    for (int v = 0; v < 8; ++v) {
      const int kk = (v < 4 ? 0 : 16) + half * 8 + (v & 3) * 2;
      up[v] = *(const unsigned*)&Ps[wave][lm][kk];
    }
    const v16bf pf = frag_cast(up);
#pragma unroll
    for (int dsub = 0; dsub < 8; ++dsub) {
      v8u uv;
#pragma unroll
      for (int v = 0; v < 8; ++v) {
        const int kk = (v < 4 ? 0 : 16) + half * 8 + (v & 3) * 2;
        uv[v] = *(const unsigned*)&Vt[dsub * 16 + lm][kk];
      }
      oacc[dsub] = __builtin_amdgcn_wmma_f32_16x16x32_bf16(
          false, pf, false, frag_cast(uv), (short)0, oacc[dsub], false, false);
    }
    __syncthreads();
  }

  // ---- normalize and write bf16 [b][s][h*128+d] for the out-projection ----
#pragma unroll
  for (int dsub = 0; dsub < 8; ++dsub)
#pragma unroll
    for (int r = 0; r < 8; ++r) {
      const float o = oacc[dsub][r] / lrow[r];
      const int srow = rowbase + r;
      const int d = dsub * 16 + lm;
      attb[(((size_t)b * SEQ) + srow) * MODEL + h * HDIM + d] = f2bf(o);
    }
}

// ---------------------------------------------------------------------------
// Launch: qkv proj (+RoPE) -> flash attention -> out proj
// Workspace: q(32MB) + k(8MB) + v(8MB) + att(32MB) bf16, ~84MB total
// ---------------------------------------------------------------------------
extern "C" void kernel_launch(void* const* d_in, const int* in_sizes, int n_in,
                              void* d_out, int out_size, void* d_ws, size_t ws_size,
                              hipStream_t stream) {
  const float* x    = (const float*)d_in[0];
  const float* cosT = (const float*)d_in[2];
  const float* sinT = (const float*)d_in[3];
  const float* Wq   = (const float*)d_in[7];
  const float* Wk   = (const float*)d_in[8];
  const float* Wv   = (const float*)d_in[9];
  const float* Wo   = (const float*)d_in[10];

  unsigned short* qb   = (unsigned short*)d_ws;
  unsigned short* kb   = qb + (size_t)NB * NH * SEQ * HDIM;
  unsigned short* vb   = kb + (size_t)NB * NKVH * SEQ * HDIM;
  unsigned short* attb = vb + (size_t)NB * NKVH * SEQ * HDIM;

  const dim3 blk(128);
  const int  Mrows = NB * SEQ;  // 4096

  gemm_wmma<0><<<dim3(MODEL / 64, Mrows / 64), blk, 0, stream>>>(
      x, nullptr, Wq, qb, MODEL, cosT, sinT, NH);
  gemm_wmma<0><<<dim3((NKVH * HDIM) / 64, Mrows / 64), blk, 0, stream>>>(
      x, nullptr, Wk, kb, NKVH * HDIM, cosT, sinT, NKVH);
  gemm_wmma<1><<<dim3((NKVH * HDIM) / 64, Mrows / 64), blk, 0, stream>>>(
      x, nullptr, Wv, vb, NKVH * HDIM, cosT, sinT, NKVH);

  flash_attn<<<dim3(SEQ / 64, NH, NB), blk, 0, stream>>>(qb, kb, vb, attb);

  gemm_wmma<2><<<dim3(MODEL / 64, Mrows / 64), blk, 0, stream>>>(
      nullptr, attb, Wo, (float*)d_out, MODEL, nullptr, nullptr, 0);
}
